// Mamba_MoE_layer_20753281974545
// MI455X (gfx1250) — compile-verified, matrix operand order checked by reading, not run on hardware
//
#include <hip/hip_runtime.h>
#include <math.h>

// ---------------- problem constants ----------------
#define D_MODEL 512
#define D_IN    1024
#define NSTATE  16
#define DT_RANK 32
#define KCONV   4
#define NEXP    8
#define BATCH   4
#define SEQ     1024
#define TOKENS  (BATCH * SEQ)   // 4096

// GEMM block tiling: 128x64 output per 256-thread block, K stepped by 32.
#define BM 128
#define BN 64
#define BK 32
#define LSTRIDE 48   // f16 units; 96B row stride: 32B-aligned v16h reads, banks spread

typedef __attribute__((ext_vector_type(16))) _Float16 v16h;
typedef __attribute__((ext_vector_type(8)))  _Float16 v8h;
typedef __attribute__((ext_vector_type(8)))  float    v8f;

// 128-bit payload type for the async-to-LDS builtin (signature discovered via
// compiler diagnostic: param 0 is 'int __attribute__((vector_size(16))) AS1 *')
typedef int v4i_t __attribute__((vector_size(16)));
typedef __attribute__((address_space(1))) v4i_t as1_v4i;
typedef __attribute__((address_space(3))) v4i_t as3_v4i;

// ---- CDNA5 async global->LDS (guarded: fall back to sync copy if absent) ----
#if __has_builtin(__builtin_amdgcn_global_load_async_to_lds_b128) && \
    __has_builtin(__builtin_amdgcn_s_wait_asynccnt)
#define HAVE_ASYNC_LDS 1
#else
#define HAVE_ASYNC_LDS 0
#endif

__device__ __forceinline__ void copy16_to_lds(const _Float16* g, _Float16* l) {
#if HAVE_ASYNC_LDS
    __builtin_amdgcn_global_load_async_to_lds_b128(
        (as1_v4i*)g, (as3_v4i*)l, 0, 0);
#else
    *(v8h*)l = *(const v8h*)g;
#endif
}
// Wait for the PREVIOUS stage's 3 async copies; ASYNCcnt completes in order, so
// with the next stage's 3 copies in flight we wait to <=3 instead of 0.
__device__ __forceinline__ void wait_async_stage(bool nextInFlight) {
#if HAVE_ASYNC_LDS
    if (nextInFlight) __builtin_amdgcn_s_wait_asynccnt(3);
    else              __builtin_amdgcn_s_wait_asynccnt(0);
#else
    (void)nextInFlight;
#endif
}

// ---------------- device helpers ----------------
__device__ __forceinline__ float silu_f(float v) {
    return v / (1.0f + __expf(-v));
}
__device__ __forceinline__ float softplus_f(float v) {
    return (v > 20.0f) ? v : log1pf(__expf(v));
}

// ---------------- weight convert + transpose: f32 [K,N] -> f16 [N,K] ----------
__global__ void transpose_f16_kernel(const float* __restrict__ in,
                                     _Float16* __restrict__ out,
                                     int Kd, int N) {
    int idx = blockIdx.x * blockDim.x + threadIdx.x;
    if (idx >= Kd * N) return;
    int k = idx / N;
    int n = idx % N;             // coalesced read along N
    out[(size_t)n * Kd + k] = (_Float16)in[idx];
}

// ---------------- RMSNorm: one wave per token; f32 and/or f16 outputs ---------
__global__ void rmsnorm_kernel(const float* __restrict__ x,
                               const float* __restrict__ w,
                               float* __restrict__ outf,      // nullable
                               _Float16* __restrict__ outh,   // nullable
                               int D) {
    int wave = (blockIdx.x * blockDim.x + threadIdx.x) >> 5;
    int lane = threadIdx.x & 31;
    if (wave >= TOKENS) return;
    const float* row = x + (size_t)wave * D;
    float s = 0.0f;
    for (int i = lane; i < D; i += 32) { float v = row[i]; s += v * v; }
    #pragma unroll
    for (int off = 16; off >= 1; off >>= 1) s += __shfl_xor(s, off, 32);
    float inv = rsqrtf(s / (float)D + 1e-5f);
    for (int i = lane; i < D; i += 32) {
        float v = row[i] * inv * w[i];
        if (outf) outf[(size_t)wave * D + i] = v;
        if (outh) outh[(size_t)wave * D + i] = (_Float16)v;
    }
}

// ---------------- double-buffered async-LDS f16 WMMA GEMM ----------------
// A: f16 row-major [M,K] (lda). Bt: f16 row-major [N,K] (ldbt) = pre-transposed B.
// Cf (f32, nullable, ldc): Cf (+)= rowscale[m]*act(dot + bias[n])
// Ch (f16, nullable, ldc): Ch  =  same value (for the next GEMM in the chain)
// act: 0=none, 1=silu, 2=softplus. M%128==0, N%64==0, K%32==0.
__global__ __launch_bounds__(256)
void wmma_gemm_kernel(const _Float16* __restrict__ A, int lda,
                      const _Float16* __restrict__ Bt, int ldbt,
                      float* __restrict__ Cf,
                      _Float16* __restrict__ Ch, int ldc,
                      const float* __restrict__ bias,
                      const float* __restrict__ rowscale, int rs_stride,
                      int M, int N, int Kdim,
                      int act, int accumulate) {
    __shared__ alignas(32) _Float16 As[2][BM * LSTRIDE];   // 2 x 12 KB
    __shared__ alignas(32) _Float16 Bs[2][BN * LSTRIDE];   // 2 x  6 KB ([n][k])

    const int blockM = blockIdx.x;
    const int blockN = blockIdx.y;
    const int tid    = threadIdx.x;
    const int w      = tid >> 5;        // wave id 0..7
    const int lane   = tid & 31;
    const int sub    = lane & 15;
    const int halfSel = lane >> 4;      // K half (0: 0..15, 1: 16..31)

    const size_t aRowBase = (size_t)(blockM * BM);
    const size_t bRowBase = (size_t)(blockN * BN);

    // staging assignments (16B units): A = 512 units, B = 256 units
    const int uA0r = tid >> 2,          uA0c = tid & 3;         // units 0..255
    const int uA1r = (tid + 256) >> 2,  uA1c = tid & 3;         // units 256..511
    const int uBr  = tid >> 2,          uBc  = tid & 3;

    #define STAGE(buf, kk)                                                       \
        do {                                                                     \
            copy16_to_lds(A  + (aRowBase + uA0r) * lda  + (kk) + uA0c * 8,       \
                          &As[buf][uA0r * LSTRIDE + uA0c * 8]);                  \
            copy16_to_lds(A  + (aRowBase + uA1r) * lda  + (kk) + uA1c * 8,       \
                          &As[buf][uA1r * LSTRIDE + uA1c * 8]);                  \
            copy16_to_lds(Bt + (bRowBase + uBr)  * ldbt + (kk) + uBc * 8,        \
                          &Bs[buf][uBr * LSTRIDE + uBc * 8]);                    \
        } while (0)

    v8f acc[4];
    #pragma unroll
    for (int nt = 0; nt < 4; ++nt) acc[nt] = (v8f){};

    // prologue: stage chunk 0 into buffer 0
    STAGE(0, 0);
    int cur = 0;

    for (int k0 = 0; k0 < Kdim; k0 += BK) {
        const bool hasNext = (k0 + BK < Kdim);
        if (hasNext) {
            STAGE(cur ^ 1, k0 + BK);            // issue next stage before waiting
            if (k0 + 2 * BK < Kdim)             // L2 prefetch two stages ahead
                __builtin_prefetch(A + (aRowBase + uA0r) * lda + k0 + 2 * BK, 0, 1);
        }
        wait_async_stage(hasNext);              // current stage landed in LDS
        __syncthreads();

        // ---- fragments + 4x WMMA (A fragment reused across 4 N-subtiles)
        const v16h a = *reinterpret_cast<const v16h*>(
            &As[cur][(w * 16 + sub) * LSTRIDE + halfSel * 16]);
        #pragma unroll
        for (int nt = 0; nt < 4; ++nt) {
            const v16h b = *reinterpret_cast<const v16h*>(
                &Bs[cur][(nt * 16 + sub) * LSTRIDE + halfSel * 16]);
            acc[nt] = __builtin_amdgcn_wmma_f32_16x16x32_f16(
                false, a, false, b, (short)0, acc[nt], false, false);
        }
        __syncthreads();                        // done reading buf[cur]
        cur ^= 1;
    }
    #undef STAGE

    // ---- epilogue: lane covers n = sub; VGPR j covers m = j + 8*halfSel
    const int mb = blockM * BM + w * 16 + halfSel * 8;
    #pragma unroll
    for (int nt = 0; nt < 4; ++nt) {
        const int n  = blockN * BN + nt * 16 + sub;
        const float bv = bias ? bias[n] : 0.0f;
        #pragma unroll
        for (int j = 0; j < 8; ++j) {
            const int m = mb + j;
            float v = acc[nt][j] + bv;
            if (act == 1)      v = silu_f(v);
            else if (act == 2) v = softplus_f(v);
            if (rowscale) v *= rowscale[(size_t)m * rs_stride];
            if (Cf) {
                float* cp = Cf + (size_t)m * ldc + n;
                if (accumulate) *cp += v; else *cp = v;
            }
            if (Ch) Ch[(size_t)m * ldc + n] = (_Float16)v;
        }
    }
}

// ---------------- Depthwise causal conv (K=4) + SiLU; f32 + f16 out ----------
__global__ void conv_silu_kernel(const float* __restrict__ xr,
                                 const float* __restrict__ cw,
                                 const float* __restrict__ cb,
                                 float* __restrict__ outf,
                                 _Float16* __restrict__ outh) {
    int tid = blockIdx.x * blockDim.x + threadIdx.x;
    if (tid >= TOKENS * D_IN) return;
    int t = tid / D_IN;
    int d = tid % D_IN;
    int l = t % SEQ;
    float acc = cb[d];
    #pragma unroll
    for (int k = 0; k < KCONV; ++k) {
        int lk = l - (KCONV - 1) + k;
        if (lk >= 0)
            acc += cw[d * KCONV + k] * xr[(size_t)(t - (KCONV - 1) + k) * (2 * D_IN) + d];
    }
    float v = silu_f(acc);
    outf[(size_t)t * D_IN + d] = v;
    outh[(size_t)t * D_IN + d] = (_Float16)v;
}

// ---------------- Selective scan: one lane per (batch, channel) --------------
__global__ void scan_kernel(const float* __restrict__ delta,
                            const float* __restrict__ u,      // xsc (f32)
                            const float* __restrict__ xdbl,   // [T,64] f32
                            const float* __restrict__ A_log,
                            const float* __restrict__ Dp,
                            const float* __restrict__ xr,     // res = xr[.,D_IN:]
                            _Float16* __restrict__ yh) {      // f16: next GEMM A
    int tid = blockIdx.x * blockDim.x + threadIdx.x;
    if (tid >= BATCH * D_IN) return;
    int b = tid / D_IN;
    int d = tid % D_IN;
    float Arow[NSTATE];
    #pragma unroll
    for (int n = 0; n < NSTATE; ++n) Arow[n] = -__expf(A_log[d * NSTATE + n]);
    const float Dd = Dp[d];
    float h[NSTATE];
    #pragma unroll
    for (int n = 0; n < NSTATE; ++n) h[n] = 0.0f;

    for (int l = 0; l < SEQ; ++l) {
        int t = b * SEQ + l;
        float dlt = delta[(size_t)t * D_IN + d];
        float uv  = u[(size_t)t * D_IN + d];
        float du  = dlt * uv;
        const float* xd = xdbl + (size_t)t * (DT_RANK + 2 * NSTATE);
        float yv = 0.0f;
        #pragma unroll
        for (int n = 0; n < NSTATE; ++n) {
            float bm = xd[DT_RANK + n];
            float cm = xd[DT_RANK + NSTATE + n];
            h[n] = __expf(dlt * Arow[n]) * h[n] + du * bm;
            yv += h[n] * cm;
        }
        float res = xr[(size_t)t * (2 * D_IN) + D_IN + d];
        yh[(size_t)t * D_IN + d] = (_Float16)((yv + uv * Dd) * silu_f(res));
    }
}

// ---------------- MoE gate: logits(512->8) + softmax ----------------
__global__ void gate_kernel(const float* __restrict__ x,
                            const float* __restrict__ Wg,
                            float* __restrict__ gates) {
    int t = blockIdx.x * blockDim.x + threadIdx.x;
    if (t >= TOKENS) return;
    const float* row = x + (size_t)t * D_MODEL;
    float logit[NEXP];
    #pragma unroll
    for (int e = 0; e < NEXP; ++e) logit[e] = 0.0f;
    for (int i = 0; i < D_MODEL; ++i) {
        float xv = row[i];
        #pragma unroll
        for (int e = 0; e < NEXP; ++e) logit[e] += xv * Wg[i * NEXP + e];
    }
    float mx = logit[0];
    #pragma unroll
    for (int e = 1; e < NEXP; ++e) mx = fmaxf(mx, logit[e]);
    float s = 0.0f;
    #pragma unroll
    for (int e = 0; e < NEXP; ++e) { logit[e] = __expf(logit[e] - mx); s += logit[e]; }
    float inv = 1.0f / s;
    #pragma unroll
    for (int e = 0; e < NEXP; ++e) gates[(size_t)t * NEXP + e] = logit[e] * inv;
}

// ---------------- host-side helpers ----------------
static inline void launch_gemm(const _Float16* A, int lda, const _Float16* Bt, int ldbt,
                               float* Cf, _Float16* Ch, int ldc, const float* bias,
                               const float* rowscale, int rs_stride,
                               int M, int N, int K, int act, int accumulate,
                               hipStream_t stream) {
    dim3 grid(M / BM, N / BN);
    wmma_gemm_kernel<<<grid, 256, 0, stream>>>(A, lda, Bt, ldbt, Cf, Ch, ldc, bias,
                                               rowscale, rs_stride, M, N, K,
                                               act, accumulate);
}
static inline void launch_transpose(const float* in, _Float16* out, int K, int N,
                                    hipStream_t stream) {
    int total = K * N;
    transpose_f16_kernel<<<(total + 255) / 256, 256, 0, stream>>>(in, out, K, N);
}

extern "C" void kernel_launch(void* const* d_in, const int* in_sizes, int n_in,
                              void* d_out, int out_size, void* d_ws, size_t ws_size,
                              hipStream_t stream) {
    (void)in_sizes; (void)n_in; (void)out_size; (void)ws_size;
    const float* inputs = (const float*)d_in[0];   // [B,L,512]
    const float* w_rms1 = (const float*)d_in[1];
    const float* w_rms2 = (const float*)d_in[2];
    const float* W_in   = (const float*)d_in[3];   // [512,2048]
    const float* conv_w = (const float*)d_in[4];   // [1024,1,4]
    const float* conv_b = (const float*)d_in[5];
    const float* W_x    = (const float*)d_in[6];   // [1024,64]
    const float* W_dt   = (const float*)d_in[7];   // [32,1024]
    const float* b_dt   = (const float*)d_in[8];
    const float* A_log  = (const float*)d_in[9];   // [1024,16]
    const float* Dp     = (const float*)d_in[10];
    const float* W_out  = (const float*)d_in[11];  // [1024,512]
    const float* W_gate = (const float*)d_in[12];  // [512,8]
    const float* W1     = (const float*)d_in[13];  // [8,512,1024]
    const float* b1     = (const float*)d_in[14];  // [8,1024]
    const float* W2     = (const float*)d_in[15];  // [8,1024,512]
    const float* b2     = (const float*)d_in[16];  // [8,512]
    float* out = (float*)d_out;                    // [T,512]

    // ---- workspace carve-up: f32 region then f16 region (all sizes 16B-multiples)
    float* wf = (float*)d_ws;
    float* xr    = wf;  wf += (size_t)TOKENS * 2 * D_IN;    // [T,2048]
    float* xsc   = wf;  wf += (size_t)TOKENS * D_IN;        // conv out (f32, scan u)
    float* xdbl  = wf;  wf += (size_t)TOKENS * 64;          // [T,64]
    float* delta = wf;  wf += (size_t)TOKENS * D_IN;
    float* x1    = wf;  wf += (size_t)TOKENS * D_MODEL;
    float* xn2   = wf;  wf += (size_t)TOKENS * D_MODEL;
    float* gates = wf;  wf += (size_t)TOKENS * NEXP;

    _Float16* wh = (_Float16*)wf;
    _Float16* W_inT  = wh;  wh += (size_t)(2 * D_IN) * D_MODEL;   // [2048,512]
    _Float16* W_xT   = wh;  wh += (size_t)64 * D_IN;              // [64,1024]
    _Float16* W_dtT  = wh;  wh += (size_t)D_IN * DT_RANK;         // [1024,32]
    _Float16* W_outT = wh;  wh += (size_t)D_MODEL * D_IN;         // [512,1024]
    _Float16* W1T    = wh;  wh += (size_t)NEXP * D_IN * D_MODEL;  // [8,1024,512]
    _Float16* W2T    = wh;  wh += (size_t)NEXP * D_MODEL * D_IN;  // [8,512,1024]
    _Float16* xn1h   = wh;  wh += (size_t)TOKENS * D_MODEL;
    _Float16* xsch   = wh;  wh += (size_t)TOKENS * D_IN;
    _Float16* xdblh  = wh;  wh += (size_t)TOKENS * 64;
    _Float16* ybh    = wh;  wh += (size_t)TOKENS * D_IN;
    _Float16* xn2h   = wh;  wh += (size_t)TOKENS * D_MODEL;
    _Float16* hbufh  = wh;  wh += (size_t)TOKENS * D_IN;

    // 0) weight convert + transpose to f16 [N,K]
    launch_transpose(W_in,  W_inT,  D_MODEL, 2 * D_IN, stream);
    launch_transpose(W_x,   W_xT,   D_IN, 64, stream);
    launch_transpose(W_dt,  W_dtT,  DT_RANK, D_IN, stream);
    launch_transpose(W_out, W_outT, D_IN, D_MODEL, stream);
    for (int e = 0; e < NEXP; ++e) {
        launch_transpose(W1 + (size_t)e * D_MODEL * D_IN,
                         W1T + (size_t)e * D_IN * D_MODEL, D_MODEL, D_IN, stream);
        launch_transpose(W2 + (size_t)e * D_IN * D_MODEL,
                         W2T + (size_t)e * D_MODEL * D_IN, D_IN, D_MODEL, stream);
    }

    // 1) RMSNorm #1 -> xn1h (f16 only; next consumer is a GEMM)
    rmsnorm_kernel<<<(TOKENS * 32 + 255) / 256, 256, 0, stream>>>(
        inputs, w_rms1, nullptr, xn1h, D_MODEL);

    // 2) xr = xn1 @ W_in   [4096x512 * 512x2048] (f32 out for conv + res gate)
    launch_gemm(xn1h, D_MODEL, W_inT, D_MODEL, xr, nullptr, 2 * D_IN,
                nullptr, nullptr, 0, TOKENS, 2 * D_IN, D_MODEL, 0, 0, stream);

    // 3) depthwise causal conv + silu -> xsc (f32) + xsch (f16)
    conv_silu_kernel<<<(TOKENS * D_IN + 255) / 256, 256, 0, stream>>>(
        xr, conv_w, conv_b, xsc, xsch);

    // 4) x_dbl = xsc @ W_x  -> xdbl (f32 for scan) + xdblh (f16 for delta GEMM)
    launch_gemm(xsch, D_IN, W_xT, D_IN, xdbl, xdblh, 64,
                nullptr, nullptr, 0, TOKENS, 64, D_IN, 0, 0, stream);

    // 5) delta = softplus(dt @ W_dt + b_dt)  (K=32: first DT_RANK cols of xdblh)
    launch_gemm(xdblh, 64, W_dtT, DT_RANK, delta, nullptr, D_IN,
                b_dt, nullptr, 0, TOKENS, D_IN, DT_RANK, 2, 0, stream);

    // 6) selective scan fused with silu(res) gate -> ybh (f16)
    scan_kernel<<<(BATCH * D_IN + 255) / 256, 256, 0, stream>>>(
        delta, xsc, xdbl, A_log, Dp, xr, ybh);

    // 7) x1 = inputs + yb @ W_out
    (void)hipMemcpyAsync(x1, inputs, (size_t)TOKENS * D_MODEL * sizeof(float),
                         hipMemcpyDeviceToDevice, stream);
    launch_gemm(ybh, D_IN, W_outT, D_IN, x1, nullptr, D_MODEL,
                nullptr, nullptr, 0, TOKENS, D_MODEL, D_IN, 0, 1, stream);

    // 8) RMSNorm #2 -> xn2 (f32 for gate) + xn2h (f16 for MoE GEMMs); gate softmax
    rmsnorm_kernel<<<(TOKENS * 32 + 255) / 256, 256, 0, stream>>>(
        x1, w_rms2, xn2, xn2h, D_MODEL);
    gate_kernel<<<(TOKENS + 255) / 256, 256, 0, stream>>>(xn2, W_gate, gates);

    // 9) out = x1 + sum_e g[:,e] * (silu(xn2@W1[e]+b1[e]) @ W2[e] + b2[e])
    (void)hipMemcpyAsync(out, x1, (size_t)TOKENS * D_MODEL * sizeof(float),
                         hipMemcpyDeviceToDevice, stream);
    for (int e = 0; e < NEXP; ++e) {
        const _Float16* W1Te = W1T + (size_t)e * D_IN * D_MODEL;
        const _Float16* W2Te = W2T + (size_t)e * D_MODEL * D_IN;
        const float* b1e = b1 + (size_t)e * D_IN;
        const float* b2e = b2 + (size_t)e * D_MODEL;
        launch_gemm(xn2h, D_MODEL, W1Te, D_MODEL, nullptr, hbufh, D_IN,
                    b1e, nullptr, 0, TOKENS, D_IN, D_MODEL, 1, 0, stream);
        launch_gemm(hbufh, D_IN, W2Te, D_IN, out, nullptr, D_MODEL,
                    b2e, gates + e, NEXP, TOKENS, D_MODEL, D_IN, 0, 1, stream);
    }
}
